// Attention_64166811402451
// MI455X (gfx1250) — matrix operand order repaired and checked by reading, not compiled, not run on hardware
//
#include <hip/hip_runtime.h>
#include <hip/hip_bf16.h>

// ---------------- types for WMMA ----------------
typedef __attribute__((ext_vector_type(16))) __bf16 v16bf;
typedef __attribute__((ext_vector_type(8)))  float  v8f;

union FragAB {
    v16bf bf;
    unsigned int u[8];
};

// native HW convert (v_cvt_pk_bf16_f32 when paired)
__device__ __forceinline__ unsigned short f2bf(float f) {
    __bf16 h = (__bf16)f;
    return __builtin_bit_cast(unsigned short, h);
}
__device__ __forceinline__ unsigned int f2bf2(float lo, float hi) {
    return (unsigned int)f2bf(lo) | ((unsigned int)f2bf(hi) << 16);
}

// A-fragment (16x32 bf16): row-major source, 16B-aligned rows -> 2x b128
__device__ __forceinline__ void loadFragA(FragAB& a, const unsigned short* row, int hh) {
    uint4 t0 = *(const uint4*)(row + hh * 8);
    uint4 t1 = *(const uint4*)(row + 16 + hh * 8);
    a.u[0] = t0.x; a.u[1] = t0.y; a.u[2] = t0.z; a.u[3] = t0.w;
    a.u[4] = t1.x; a.u[5] = t1.y; a.u[6] = t1.z; a.u[7] = t1.w;
}
// B-fragment (32x16 bf16): "row" = the n-th row of the K-major tile -> 2x b128
__device__ __forceinline__ void loadFragB(FragAB& b, const unsigned short* row, int hh) {
    uint4 t0 = *(const uint4*)(row + hh * 8);
    uint4 t1 = *(const uint4*)(row + 16 + hh * 8);
    b.u[0] = t0.x; b.u[1] = t0.y; b.u[2] = t0.z; b.u[3] = t0.w;
    b.u[4] = t1.x; b.u[5] = t1.y; b.u[6] = t1.z; b.u[7] = t1.w;
}

#define DMODEL 768
#define NHEADS 12
#define DHEAD  64
#define BATCH  2
#define SEQ    2048
#define NTOK   (BATCH * SEQ)      // 4096
#define QKVDIM (3 * DMODEL)       // 2304

#define APAD 40    // 80 B rows: 16B aligned, 20-bank stride (conflict-free)
#define KPAD 72    // 144 B rows: 16B aligned, 36-bank stride (conflict-free)

// =====================================================================
// Kernel 1: qkv = x @ W_qkv^T + b_qkv  ->  bf16 ws [which][b][h][s][64]
// grid (NTOK/128, QKVDIM/64), block 256 (8 waves). Double-buffered LDS.
// =====================================================================
__global__ __launch_bounds__(256)
void qkv_proj_kernel(const float* __restrict__ x,
                     const float* __restrict__ Wqkv,
                     const float* __restrict__ bqkv,
                     unsigned short* __restrict__ qkv) {
    __shared__ __align__(16) unsigned short As[2][128][APAD];
    __shared__ __align__(16) unsigned short Bs[2][64][APAD];

    const int tid = threadIdx.x;
    const int wave = tid >> 5;
    const int lane = tid & 31;
    const int lm = lane & 15;
    const int hh = lane >> 4;
    const int rowBase = blockIdx.x * 128;
    const int nBase   = blockIdx.y * 64;

    float4 xa[4], wb[2];

    auto loadRegs = [&](int kt) {
        const int k0 = kt * 32;
        #pragma unroll
        for (int p = 0; p < 4; ++p) {
            int c = tid + p * 256, row = c >> 3, col4 = (c & 7) * 4;
            xa[p] = *(const float4*)(x + (size_t)(rowBase + row) * DMODEL + k0 + col4);
        }
        #pragma unroll
        for (int p = 0; p < 2; ++p) {
            int c = tid + p * 256, row = c >> 3, col4 = (c & 7) * 4;
            wb[p] = *(const float4*)(Wqkv + (size_t)(nBase + row) * DMODEL + k0 + col4);
        }
    };
    auto storeTile = [&](int buf) {
        #pragma unroll
        for (int p = 0; p < 4; ++p) {
            int c = tid + p * 256, row = c >> 3, col4 = (c & 7) * 4;
            uint2 pk = { f2bf2(xa[p].x, xa[p].y), f2bf2(xa[p].z, xa[p].w) };
            *(uint2*)&As[buf][row][col4] = pk;
        }
        #pragma unroll
        for (int p = 0; p < 2; ++p) {
            int c = tid + p * 256, row = c >> 3, col4 = (c & 7) * 4;
            uint2 pk = { f2bf2(wb[p].x, wb[p].y), f2bf2(wb[p].z, wb[p].w) };
            *(uint2*)&Bs[buf][row][col4] = pk;
        }
    };

    v8f acc[4] = {};

    loadRegs(0);
    storeTile(0);
    __syncthreads();

    for (int kt = 0; kt < DMODEL / 32; ++kt) {
        const int cur = kt & 1;
        if (kt + 1 < DMODEL / 32) loadRegs(kt + 1);   // global loads in flight

        FragAB a;
        loadFragA(a, &As[cur][wave * 16 + lm][0], hh);
        FragAB bb[4];
        #pragma unroll
        for (int f = 0; f < 4; ++f)
            loadFragB(bb[f], &Bs[cur][f * 16 + lm][0], hh);
        #pragma unroll
        for (int f = 0; f < 4; ++f)
            acc[f] = __builtin_amdgcn_wmma_f32_16x16x32_bf16(
                false, a.bf, false, bb[f].bf, (short)0, acc[f], false, false);

        if (kt + 1 < DMODEL / 32) storeTile((kt + 1) & 1);
        __syncthreads();
    }

    // ---- epilogue: uniform (scalar) index math per block ----
    const int which = nBase / DMODEL;                 // 0..2, uniform
    const int head  = (nBase - which * DMODEL) >> 6;  // uniform
    const int b2    = rowBase >> 11;                  // uniform
    const int sBase = (rowBase & (SEQ - 1)) + wave * 16;
    const size_t outBase =
        ((((size_t)which * BATCH + b2) * NHEADS + head) * SEQ + sBase) * DHEAD;

    float bias[4];
    #pragma unroll
    for (int f = 0; f < 4; ++f) bias[f] = bqkv[nBase + f * 16 + lm];

    #pragma unroll
    for (int f = 0; f < 4; ++f)
        #pragma unroll
        for (int r = 0; r < 8; ++r)
            qkv[outBase + (size_t)(r + hh * 8) * DHEAD + f * 16 + lm] =
                f2bf(acc[f][r] + bias[f]);
}

// =====================================================================
// Kernel 2: causal flash attention.  grid (SEQ/64, NHEADS, BATCH), block 128.
// Double-buffered K/V LDS; 1 barrier per kv block; exp2-domain softmax.
// =====================================================================
__global__ __launch_bounds__(128)
void attn_kernel(const unsigned short* __restrict__ qkv,
                 unsigned short* __restrict__ z) {
    __shared__ __align__(16) unsigned short Ks[2][32][KPAD];  // keys row-major
    __shared__ __align__(16) unsigned short Vt[2][64][APAD];  // V transposed
    __shared__ __align__(16) unsigned short Ps[4][16][APAD];  // per-wave P

    const int tid = threadIdx.x;
    const int wave = tid >> 5;
    const int lane = tid & 31;
    const int lm = lane & 15;
    const int hh = lane >> 4;
    const int b = blockIdx.z;
    const int h = blockIdx.y;

    const size_t headStride = (size_t)SEQ * DHEAD;
    const unsigned short* qh = qkv + (((size_t)0 * BATCH + b) * NHEADS + h) * headStride;
    const unsigned short* kh = qkv + (((size_t)1 * BATCH + b) * NHEADS + h) * headStride;
    const unsigned short* vh = qkv + (((size_t)2 * BATCH + b) * NHEADS + h) * headStride;

    const int q0 = blockIdx.x * 64 + wave * 16;

    // log2-domain scale: score * (1/8) * log2(e)
    const float kScale = 0.125f * 1.44269504088896340736f;

    // Q fragments resident for whole kv loop (4x global_load_b128)
    FragAB aq0, aq1;
    {
        const unsigned short* qrow = qh + (size_t)(q0 + lm) * DHEAD;
        loadFragA(aq0, qrow, hh);
        loadFragA(aq1, qrow + 32, hh);
    }

    uint4 kr[2], vr[2];
    auto kvLoad = [&](int kv0) {
        #pragma unroll
        for (int p = 0; p < 2; ++p) {
            int c = tid + p * 128, row = c >> 3, col8 = (c & 7) * 8;
            kr[p] = *(const uint4*)(kh + (size_t)(kv0 + row) * DHEAD + col8);
            vr[p] = *(const uint4*)(vh + (size_t)(kv0 + row) * DHEAD + col8);
        }
    };
    auto kvStore = [&](int buf) {
        #pragma unroll
        for (int p = 0; p < 2; ++p) {
            int c = tid + p * 128, row = c >> 3, col8 = (c & 7) * 8;
            *(uint4*)&Ks[buf][row][col8] = kr[p];               // ds_store_b128
            unsigned int vw[4] = { vr[p].x, vr[p].y, vr[p].z, vr[p].w };
            #pragma unroll
            for (int i = 0; i < 4; ++i) {
                Vt[buf][col8 + 2 * i][row]     = (unsigned short)(vw[i] & 0xFFFFu);
                Vt[buf][col8 + 2 * i + 1][row] = (unsigned short)(vw[i] >> 16);
            }
        }
    };

    v8f o[4] = {};
    float rm[8], rl[8];
    #pragma unroll
    for (int r = 0; r < 8; ++r) { rm[r] = -__builtin_inff(); rl[r] = 0.0f; }

    const int nkv = (blockIdx.x + 1) * 2;   // kv blocks of 32 (causal bound)

    kvLoad(0);
    kvStore(0);
    __syncthreads();

    for (int kb = 0; kb < nkv; ++kb) {
        const int cur = kb & 1;
        const int kv0 = kb * 32;
        if (kb + 1 < nkv) kvLoad((kb + 1) * 32);    // next block in flight

        // ---- scores: two 16-key sub-tiles, each d-split over 2 WMMA ----
        v8f s0f, s1f;
        #pragma unroll
        for (int sub = 0; sub < 2; ++sub) {
            const unsigned short* krow = &Ks[cur][sub * 16 + lm][0];
            FragAB bk0, bk1;
            loadFragB(bk0, krow, hh);
            loadFragB(bk1, krow + 32, hh);
            v8f zero = {};
            v8f sc = __builtin_amdgcn_wmma_f32_16x16x32_bf16(
                false, aq0.bf, false, bk0.bf, (short)0, zero, false, false);
            sc = __builtin_amdgcn_wmma_f32_16x16x32_bf16(
                false, aq1.bf, false, bk1.bf, (short)0, sc, false, false);
            if (sub == 0) s0f = sc; else s1f = sc;
        }

        // ---- online softmax in log2 domain ----
        #pragma unroll
        for (int r = 0; r < 8; ++r) {
            const int M = r + hh * 8;
            const int qi = q0 + M;
            float s0 = s0f[r] * kScale;
            float s1 = s1f[r] * kScale;
            if (kv0 + lm > qi)      s0 = -__builtin_inff();
            if (kv0 + 16 + lm > qi) s1 = -__builtin_inff();
            float mx = fmaxf(s0, s1);
            #pragma unroll
            for (int m = 8; m >= 1; m >>= 1)
                mx = fmaxf(mx, __shfl_xor(mx, m, 32));
            float nm = fmaxf(rm[r], mx);
            float al = exp2f(rm[r] - nm);
            float p0 = exp2f(s0 - nm);
            float p1 = exp2f(s1 - nm);
            float rs = p0 + p1;
            #pragma unroll
            for (int m = 8; m >= 1; m >>= 1)
                rs += __shfl_xor(rs, m, 32);
            rl[r] = al * rl[r] + rs;
            rm[r] = nm;
            #pragma unroll
            for (int f = 0; f < 4; ++f)
                o[f][r] *= al;
            Ps[wave][M][lm]      = f2bf(p0);
            Ps[wave][M][lm + 16] = f2bf(p1);
        }
        // P write -> read is same-wave LDS: DS ops are wave-ordered; compiler
        // inserts the s_wait_dscnt via the may-alias dependence on Ps.

        // ---- P (16x32) as A-fragment, 4 WMMA with V columns ----
        FragAB ap;
        loadFragA(ap, &Ps[wave][lm][0], hh);
        FragAB bv[4];
        #pragma unroll
        for (int f = 0; f < 4; ++f)
            loadFragB(bv[f], &Vt[cur][f * 16 + lm][0], hh);
        #pragma unroll
        for (int f = 0; f < 4; ++f)
            o[f] = __builtin_amdgcn_wmma_f32_16x16x32_bf16(
                false, ap.bf, false, bv[f].bf, (short)0, o[f], false, false);

        if (kb + 1 < nkv) kvStore((kb + 1) & 1);
        __syncthreads();
    }

    // ---- normalize and store z (bf16, [token][768]) ----
    #pragma unroll
    for (int r = 0; r < 8; ++r) {
        float inv = 1.0f / rl[r];
        int t = b * SEQ + q0 + r + hh * 8;
        #pragma unroll
        for (int f = 0; f < 4; ++f) {
            int cc = h * DHEAD + f * 16 + lm;
            z[(size_t)t * DMODEL + cc] = f2bf(o[f][r] * inv);
        }
    }
}

// =====================================================================
// Kernel 3: out = z @ W_o^T + b_o (fp32 out). Double-buffered LDS.
// grid (NTOK/128, DMODEL/64), block 256.
// =====================================================================
__global__ __launch_bounds__(256)
void out_proj_kernel(const unsigned short* __restrict__ z,
                     const float* __restrict__ Wo,
                     const float* __restrict__ bo,
                     float* __restrict__ out) {
    __shared__ __align__(16) unsigned short As[2][128][APAD];
    __shared__ __align__(16) unsigned short Bs[2][64][APAD];

    const int tid = threadIdx.x;
    const int wave = tid >> 5;
    const int lane = tid & 31;
    const int lm = lane & 15;
    const int hh = lane >> 4;
    const int rowBase = blockIdx.x * 128;
    const int nBase   = blockIdx.y * 64;

    uint4  za[2];
    float4 wb[2];

    auto loadRegs = [&](int kt) {
        const int k0 = kt * 32;
        #pragma unroll
        for (int p = 0; p < 2; ++p) {
            int c = tid + p * 256, row = c >> 2, col8 = (c & 3) * 8;
            za[p] = *(const uint4*)(z + (size_t)(rowBase + row) * DMODEL + k0 + col8);
        }
        #pragma unroll
        for (int p = 0; p < 2; ++p) {
            int c = tid + p * 256, row = c >> 3, col4 = (c & 7) * 4;
            wb[p] = *(const float4*)(Wo + (size_t)(nBase + row) * DMODEL + k0 + col4);
        }
    };
    auto storeTile = [&](int buf) {
        #pragma unroll
        for (int p = 0; p < 2; ++p) {
            int c = tid + p * 256, row = c >> 2, col8 = (c & 3) * 8;
            *(uint4*)&As[buf][row][col8] = za[p];               // ds_store_b128
        }
        #pragma unroll
        for (int p = 0; p < 2; ++p) {
            int c = tid + p * 256, row = c >> 3, col4 = (c & 7) * 4;
            uint2 pk = { f2bf2(wb[p].x, wb[p].y), f2bf2(wb[p].z, wb[p].w) };
            *(uint2*)&Bs[buf][row][col4] = pk;
        }
    };

    v8f acc[4] = {};

    loadRegs(0);
    storeTile(0);
    __syncthreads();

    for (int kt = 0; kt < DMODEL / 32; ++kt) {
        const int cur = kt & 1;
        if (kt + 1 < DMODEL / 32) loadRegs(kt + 1);

        FragAB a;
        loadFragA(a, &As[cur][wave * 16 + lm][0], hh);
        FragAB bb[4];
        #pragma unroll
        for (int f = 0; f < 4; ++f)
            loadFragB(bb[f], &Bs[cur][f * 16 + lm][0], hh);
        #pragma unroll
        for (int f = 0; f < 4; ++f)
            acc[f] = __builtin_amdgcn_wmma_f32_16x16x32_bf16(
                false, a.bf, false, bb[f].bf, (short)0, acc[f], false, false);

        if (kt + 1 < DMODEL / 32) storeTile((kt + 1) & 1);
        __syncthreads();
    }

    float bias[4];
    #pragma unroll
    for (int f = 0; f < 4; ++f) bias[f] = bo[nBase + f * 16 + lm];

    #pragma unroll
    for (int f = 0; f < 4; ++f)
        #pragma unroll
        for (int r = 0; r < 8; ++r)
            out[(size_t)(rowBase + wave * 16 + r + hh * 8) * DMODEL + nBase + f * 16 + lm] =
                acc[f][r] + bias[f];
}

// =====================================================================
extern "C" void kernel_launch(void* const* d_in, const int* in_sizes, int n_in,
                              void* d_out, int out_size, void* d_ws, size_t ws_size,
                              hipStream_t stream) {
    const float* x     = (const float*)d_in[0];
    const float* Wqkv  = (const float*)d_in[1];
    const float* bqkv  = (const float*)d_in[2];
    const float* Wo    = (const float*)d_in[3];
    const float* bo    = (const float*)d_in[4];
    float* out = (float*)d_out;

    unsigned short* qkv_ws = (unsigned short*)d_ws;                 // 3*4096*768 bf16
    unsigned short* z_ws   = qkv_ws + (size_t)3 * NTOK * DMODEL;    // 4096*768 bf16

    qkv_proj_kernel<<<dim3(NTOK / 128, QKVDIM / 64), 256, 0, stream>>>(x, Wqkv, bqkv, qkv_ws);
    attn_kernel<<<dim3(SEQ / 64, NHEADS, BATCH), 128, 0, stream>>>(qkv_ws, z_ws);
    out_proj_kernel<<<dim3(NTOK / 128, DMODEL / 64), 256, 0, stream>>>(z_ws, Wo, bo, out);
}
